// MixedOp_53807350284626
// MI455X (gfx1250) — compile-verified, matrix operand order checked by reading, not run on hardware
//
#include <hip/hip_runtime.h>
#include <math.h>

#define B_   64
#define C_   512
#define CP_  128
#define HW_  1024
#define NP_  (B_*CP_*HW_)   /* 8388608  */
#define NX_  (B_*C_*HW_)    /* 33554432 */
#define EPS_ 1e-5f

typedef __attribute__((ext_vector_type(16))) __bf16 v16bf;
typedef __attribute__((ext_vector_type(8)))  __bf16 v8bf;
typedef __attribute__((ext_vector_type(8)))  float  v8f;
typedef __attribute__((ext_vector_type(4)))  unsigned int u32x4;

static __device__ __forceinline__ unsigned short f2bf(float f) {
    unsigned int u = __float_as_uint(f);
    u += 0x7FFFu + ((u >> 16) & 1u);           // round-to-nearest-even
    return (unsigned short)(u >> 16);
}

// Load one 16x32 (A) / 32x16 (B) bf16 fragment half-row: two 16B dwordx4 loads.
// Fragment-local K for element i of lane L: 8*(L>=16) + (i<8 ? i : i+8).
static __device__ __forceinline__ v16bf ldfrag(const unsigned short* p) {
    v8bf lo = *reinterpret_cast<const v8bf*>(p);
    v8bf hi = *reinterpret_cast<const v8bf*>(p + 16);
    return __builtin_shufflevector(lo, hi, 0,1,2,3,4,5,6,7,8,9,10,11,12,13,14,15);
}
static __device__ __forceinline__ v16bf zerofrag() {
    u32x4 z = {0u,0u,0u,0u};
    v8bf zb = __builtin_bit_cast(v8bf, z);
    return __builtin_shufflevector(zb, zb, 0,1,2,3,4,5,6,7,8,9,10,11,12,13,14,15);
}
static __device__ __forceinline__ v8f wmma_bf16(v16bf a, v16bf b, v8f c) {
    return __builtin_amdgcn_wmma_f32_16x16x32_bf16(false, a, false, b, (short)0, c, false, false);
}

//====================== stage 0: spatial avg/max per (b,c) ======================
__global__ void reduce_pool_k(const float* __restrict__ x, float* __restrict__ avgb,
                              float* __restrict__ maxb) {
    int bc = blockIdx.x, tid = threadIdx.x;
    const float* p = x + ((size_t)bc << 10);
    float s = 0.f, m = -INFINITY;
    for (int hw = tid; hw < HW_; hw += 256) { float v = p[hw]; s += v; m = fmaxf(m, v); }
    __shared__ float ss[256], mm[256];
    ss[tid] = s; mm[tid] = m; __syncthreads();
    for (int st = 128; st > 0; st >>= 1) {
        if (tid < st) { ss[tid] += ss[tid+st]; mm[tid] = fmaxf(mm[tid], mm[tid+st]); }
        __syncthreads();
    }
    if (tid == 0) { avgb[bc] = ss[0] * (1.f/1024.f); maxb[bc] = mm[0]; }
}

//====================== stage 1: channel attention MLP ======================
__global__ void attn_k(const float* __restrict__ avgb, const float* __restrict__ maxb,
                       const float* __restrict__ wall,  // (14,10)
                       const float* __restrict__ wfc1,  // (10,1024)
                       const float* __restrict__ wfc2,  // (512,10)
                       float* __restrict__ ca) {
    int b = blockIdx.x, tid = threadIdx.x;
    __shared__ float A[100], y0[10], y1[10], red[256];
    if (tid < 100) {
        int i = tid / 10, j = tid % 10; float s = 0.f;
        for (int r = 0; r < 14; ++r) s += wall[r*10+i] * wall[r*10+j];
        A[tid] = s;
    }
    __syncthreads();
    for (int i = 0; i < 10; ++i) {
        float s = 0.f;
        for (int k = tid; k < 1024; k += 256) {
            float pv = (k < C_) ? avgb[b*C_ + k] : maxb[b*C_ + (k - C_)];
            s += pv * wfc1[i*1024 + k];
        }
        red[tid] = s; __syncthreads();
        for (int st = 128; st > 0; st >>= 1) { if (tid < st) red[tid] += red[tid+st]; __syncthreads(); }
        if (tid == 0) y0[i] = red[0];
        __syncthreads();
    }
    if (tid < 10) {
        float s = 0.f;
        for (int j = 0; j < 10; ++j) s += A[tid*10 + j] * y0[j];
        y1[tid] = fmaxf(s, 0.f);
    }
    __syncthreads();
    for (int c = tid; c < C_; c += 256) {
        float s = 0.f;
        for (int i = 0; i < 10; ++i) s += wfc2[c*10 + i] * y1[i];
        ca[b*C_ + c] = 1.f / (1.f + expf(-s));
    }
}

//====================== stage 2: slist + top-k ======================
__global__ void slist_k(const float* __restrict__ ca, float* __restrict__ slist) {
    int c = blockIdx.x * 256 + threadIdx.x;
    float s = 0.f;
    for (int b = 0; b < B_; ++b) s += ca[b*C_ + c];
    slist[c] = s;
}

__global__ void topk_k(const float* __restrict__ slist, int* __restrict__ idxb) {
    int tid = threadIdx.x;                       // 512 threads
    __shared__ float v[512], rv[512];
    __shared__ int picked[512], ri[512];
    v[tid] = slist[tid]; picked[tid] = 0; __syncthreads();
    for (int it = 0; it < CP_; ++it) {
        rv[tid] = picked[tid] ? -INFINITY : v[tid];
        ri[tid] = tid; __syncthreads();
        for (int st = 256; st > 0; st >>= 1) {
            if (tid < st) {
                float a = rv[tid], b = rv[tid+st]; int ia = ri[tid], ib = ri[tid+st];
                if (b > a || (b == a && ib < ia)) { rv[tid] = b; ri[tid] = ib; }
            }
            __syncthreads();
        }
        if (tid == 0) { idxb[it] = ri[0]; picked[ri[0]] = 1; }
        __syncthreads();
    }
}

//====================== stage 3: scale + gather ======================
__global__ void scale_k(const float* __restrict__ x, const float* __restrict__ ca,
                        float* __restrict__ out) {
    int i = blockIdx.x * 256 + threadIdx.x;
    int b = i >> 19, c = (i >> 10) & (C_-1);
    out[i] = x[i] * ca[b*C_ + c];
}
__global__ void gather_k(const float* __restrict__ out, const int* __restrict__ idxb,
                         float* __restrict__ xtemp) {
    int i = blockIdx.x * 256 + threadIdx.x;
    int hw = i & (HW_-1), j = (i >> 10) & (CP_-1), b = i >> 17;
    xtemp[i] = out[(((size_t)b*C_ + idxb[j]) << 10) + hw];
}

//====================== pools ======================
__global__ void maxpool3_k(const float* __restrict__ src, float* __restrict__ dst) {
    int i = blockIdx.x * 256 + threadIdx.x;
    int hw = i & (HW_-1), h = hw >> 5, w = hw & 31;
    const float* p = src + (size_t)(i & ~(HW_-1));
    float m = -INFINITY;
    for (int dy = -1; dy <= 1; ++dy) for (int dx = -1; dx <= 1; ++dx) {
        int y = h + dy, xx = w + dx;
        if (y >= 0 && y < 32 && xx >= 0 && xx < 32) m = fmaxf(m, p[(y<<5) + xx]);
    }
    dst[i] = m;
}
__global__ void avgpool3_k(const float* __restrict__ src, float* __restrict__ dst) {
    int i = blockIdx.x * 256 + threadIdx.x;
    int hw = i & (HW_-1), h = hw >> 5, w = hw & 31;
    const float* p = src + (size_t)(i & ~(HW_-1));
    float s = 0.f; int cnt = 0;
    for (int dy = -1; dy <= 1; ++dy) for (int dx = -1; dx <= 1; ++dx) {
        int y = h + dy, xx = w + dx;
        if (y >= 0 && y < 32 && xx >= 0 && xx < 32) { s += p[(y<<5) + xx]; ++cnt; }
    }
    dst[i] = s / (float)cnt;
}

//====================== BN helpers ======================
__global__ void bn_stats_k(const float* __restrict__ src, float* __restrict__ mbuf,
                           float* __restrict__ vbuf) {
    int j = blockIdx.x, tid = threadIdx.x;
    float s = 0.f, q = 0.f;
    for (int b = 0; b < B_; ++b) {
        const float* p = src + (((size_t)b*CP_ + j) << 10);
        for (int hw = tid; hw < HW_; hw += 256) { float v = p[hw]; s += v; q += v*v; }
    }
    __shared__ float ss[256], qq[256];
    ss[tid] = s; qq[tid] = q; __syncthreads();
    for (int st = 128; st > 0; st >>= 1) {
        if (tid < st) { ss[tid] += ss[tid+st]; qq[tid] += qq[tid+st]; }
        __syncthreads();
    }
    if (tid == 0) {
        float m = ss[0] * (1.f/65536.f);
        mbuf[j] = m; vbuf[j] = qq[0] * (1.f/65536.f) - m*m;
    }
}
__global__ void bnrelu_k(const float* __restrict__ src, const float* __restrict__ m,
                         const float* __restrict__ v, float* __restrict__ dst) {
    int i = blockIdx.x * 256 + threadIdx.x;
    int j = (i >> 10) & (CP_-1);
    dst[i] = fmaxf((src[i] - m[j]) * rsqrtf(v[j] + EPS_), 0.f);
}
__global__ void acc_bn_k(float* __restrict__ acc, const float* __restrict__ src,
                         const float* __restrict__ m, const float* __restrict__ v,
                         const float* __restrict__ wts, int widx) {
    int i = blockIdx.x * 256 + threadIdx.x;
    int j = (i >> 10) & (CP_-1);
    acc[i] += wts[widx] * (src[i] - m[j]) * rsqrtf(v[j] + EPS_);
}
__global__ void acc_init_k(float* __restrict__ acc, const float* __restrict__ src,
                           const float* __restrict__ wts) {
    int i = blockIdx.x * 256 + threadIdx.x;
    acc[i] = wts[3] * src[i];
}

//====================== depthwise conv -> bf16 [b][hw][c] ======================
__global__ void dwconv_bf16_k(const float* __restrict__ src, const float* __restrict__ dw,
                              unsigned short* __restrict__ dst,
                              int k, int dil, int pad, int dorelu) {
    int i = blockIdx.x * 256 + threadIdx.x;
    int hw = i & (HW_-1), j = (i >> 10) & (CP_-1), b = i >> 17;
    int h = hw >> 5, w = hw & 31;
    const float* p = src + (size_t)(i & ~(HW_-1));
    float acc = 0.f;
    for (int ky = 0; ky < k; ++ky) {
        int y = h + ky*dil - pad;
        if (y < 0 || y >= 32) continue;
        for (int kx = 0; kx < k; ++kx) {
            int xx = w + kx*dil - pad;
            if (xx < 0 || xx >= 32) continue;
            float v = p[(y<<5) + xx];
            if (dorelu) v = fmaxf(v, 0.f);
            acc += v * dw[j*k*k + ky*k + kx];
        }
    }
    dst[(((size_t)b << 10) + hw) * CP_ + j] = f2bf(acc);
}

//====================== weight / activation converters ======================
__global__ void cvt_pw_k(const float* __restrict__ src, unsigned short* __restrict__ dst) {
    int i = blockIdx.x * 256 + threadIdx.x;      // CP*CP
    dst[i] = f2bf(src[i]);
}
__global__ void cvt_tap_k(const float* __restrict__ src, unsigned short* __restrict__ dst) {
    int i = blockIdx.x * 256 + threadIdx.x;      // CP*CP*7, src flat (co,ci,t)
    int co = i / (CP_*7), rem = i % (CP_*7), ci = rem / 7, t = rem % 7;
    dst[(t*CP_ + co) * CP_ + ci] = f2bf(src[i]);
}
__global__ void relu_cvt_k(const float* __restrict__ src, unsigned short* __restrict__ dst) {
    int i = blockIdx.x * 256 + threadIdx.x;      // NCHW index
    int hw = i & (HW_-1), j = (i >> 10) & (CP_-1), b = i >> 17;
    dst[(((size_t)b << 10) + hw) * CP_ + j] = f2bf(fmaxf(src[i], 0.f));
}

//====================== WMMA GEMM: pointwise conv ======================
// Out[b][co][hw] = sum_ci W[co][ci] * X[b][hw][ci].
// One wave = 16(M) x 64(N) macro-tile: A fragments held in registers, 16 WMMAs.
__global__ void pw_gemm_k(const unsigned short* __restrict__ Xb,
                          const unsigned short* __restrict__ Wb,
                          float* __restrict__ Out) {
    int lane = threadIdx.x;
    int n0 = blockIdx.x << 6, m0 = blockIdx.y << 4, b = blockIdx.z;
    int lr = lane & 15, half = lane >> 4, base8 = half << 3;
    const unsigned short* arow = Wb + (m0 + lr) * CP_ + base8;
    v16bf a0 = ldfrag(arow +  0), a1 = ldfrag(arow + 32);
    v16bf a2 = ldfrag(arow + 64), a3 = ldfrag(arow + 96);
    #pragma unroll
    for (int nt = 0; nt < 4; ++nt) {
        int col = n0 + (nt << 4) + lr;
        const unsigned short* brow = Xb + (((size_t)b << 10) + col) * CP_ + base8;
        v8f c = {};
        c = wmma_bf16(a0, ldfrag(brow +  0), c);
        c = wmma_bf16(a1, ldfrag(brow + 32), c);
        c = wmma_bf16(a2, ldfrag(brow + 64), c);
        c = wmma_bf16(a3, ldfrag(brow + 96), c);
        float* o = Out + (((size_t)b*CP_ + m0 + (half << 3)) << 10) + col;
        #pragma unroll
        for (int r = 0; r < 8; ++r) o[(size_t)r << 10] = c[r];
    }
}

//====================== WMMA GEMM: 1x7 conv (taps over w), bf16 out ======================
__global__ void gemm1x7_k(const unsigned short* __restrict__ Xb,
                          const unsigned short* __restrict__ W7,
                          unsigned short* __restrict__ Ob) {
    int lane = threadIdx.x;
    int n0 = blockIdx.x << 6, m0 = blockIdx.y << 4, b = blockIdx.z;
    int lr = lane & 15, half = lane >> 4, base8 = half << 3;
    v8f c[4] = {{}, {}, {}, {}};
    for (int t = 0; t < 7; ++t) {
        const unsigned short* arow = W7 + ((size_t)(t*CP_ + m0 + lr)) * CP_ + base8;
        v16bf a0 = ldfrag(arow +  0), a1 = ldfrag(arow + 32);
        v16bf a2 = ldfrag(arow + 64), a3 = ldfrag(arow + 96);
        #pragma unroll
        for (int nt = 0; nt < 4; ++nt) {
            int hw = n0 + (nt << 4) + lr, w = hw & 31;
            int wp = w + t - 3;
            bool valid = (wp >= 0) && (wp < 32);
            int hwp = (hw & ~31) | (wp & 31);
            const unsigned short* brow = Xb + (((size_t)b << 10) + hwp) * CP_ + base8;
            c[nt] = wmma_bf16(a0, valid ? ldfrag(brow +  0) : zerofrag(), c[nt]);
            c[nt] = wmma_bf16(a1, valid ? ldfrag(brow + 32) : zerofrag(), c[nt]);
            c[nt] = wmma_bf16(a2, valid ? ldfrag(brow + 64) : zerofrag(), c[nt]);
            c[nt] = wmma_bf16(a3, valid ? ldfrag(brow + 96) : zerofrag(), c[nt]);
        }
    }
    #pragma unroll
    for (int nt = 0; nt < 4; ++nt) {
        int hw = n0 + (nt << 4) + lr;
        unsigned short* o = Ob + (((size_t)b << 10) + hw) * CP_ + m0 + (half << 3);
        #pragma unroll
        for (int r = 0; r < 8; ++r) o[r] = f2bf(c[nt][r]);
    }
}

//====================== WMMA GEMM: 7x1 conv (taps over h), f32 NCHW out ======================
__global__ void gemm7x1_k(const unsigned short* __restrict__ Xb,
                          const unsigned short* __restrict__ W7,
                          float* __restrict__ Out) {
    int lane = threadIdx.x;
    int n0 = blockIdx.x << 6, m0 = blockIdx.y << 4, b = blockIdx.z;
    int lr = lane & 15, half = lane >> 4, base8 = half << 3;
    v8f c[4] = {{}, {}, {}, {}};
    for (int t = 0; t < 7; ++t) {
        const unsigned short* arow = W7 + ((size_t)(t*CP_ + m0 + lr)) * CP_ + base8;
        v16bf a0 = ldfrag(arow +  0), a1 = ldfrag(arow + 32);
        v16bf a2 = ldfrag(arow + 64), a3 = ldfrag(arow + 96);
        #pragma unroll
        for (int nt = 0; nt < 4; ++nt) {
            int hw = n0 + (nt << 4) + lr, h = hw >> 5;
            int hp = h + t - 3;
            bool valid = (hp >= 0) && (hp < 32);
            int hwp = ((hp & 31) << 5) | (hw & 31);
            const unsigned short* brow = Xb + (((size_t)b << 10) + hwp) * CP_ + base8;
            c[nt] = wmma_bf16(a0, valid ? ldfrag(brow +  0) : zerofrag(), c[nt]);
            c[nt] = wmma_bf16(a1, valid ? ldfrag(brow + 32) : zerofrag(), c[nt]);
            c[nt] = wmma_bf16(a2, valid ? ldfrag(brow + 64) : zerofrag(), c[nt]);
            c[nt] = wmma_bf16(a3, valid ? ldfrag(brow + 96) : zerofrag(), c[nt]);
        }
    }
    #pragma unroll
    for (int nt = 0; nt < 4; ++nt) {
        int col = n0 + (nt << 4) + lr;
        float* o = Out + (((size_t)b*CP_ + m0 + (half << 3)) << 10) + col;
        #pragma unroll
        for (int r = 0; r < 8; ++r) o[(size_t)r << 10] = c[nt][r];
    }
}

//====================== final scatter ======================
__global__ void scatter_k(const float* __restrict__ acc, const int* __restrict__ idxb,
                          float* __restrict__ out) {
    int i = blockIdx.x * 256 + threadIdx.x;
    int hw = i & (HW_-1), j = (i >> 10) & (CP_-1), b = i >> 17;
    out[(((size_t)b*C_ + idxb[j]) << 10) + hw] = acc[i];
}

//====================================================================
extern "C" void kernel_launch(void* const* d_in, const int* in_sizes, int n_in,
                              void* d_out, int out_size, void* d_ws, size_t ws_size,
                              hipStream_t stream) {
    (void)in_sizes; (void)n_in; (void)out_size; (void)ws_size;
    const float* x    = (const float*)d_in[0];
    const float* wts  = (const float*)d_in[1];   // (10,)
    const float* wall = (const float*)d_in[2];   // (14,10)
    const float* wfc1 = (const float*)d_in[3];   // (10,1024)
    const float* wfc2 = (const float*)d_in[4];   // (512,10)
    const float* sep_dw1[3] = {(const float*)d_in[5],  (const float*)d_in[9],  (const float*)d_in[13]};
    const float* sep_pw1[3] = {(const float*)d_in[6],  (const float*)d_in[10], (const float*)d_in[14]};
    const float* sep_dw2[3] = {(const float*)d_in[7],  (const float*)d_in[11], (const float*)d_in[15]};
    const float* sep_pw2[3] = {(const float*)d_in[8],  (const float*)d_in[12], (const float*)d_in[16]};
    const float* dil_dw[2]  = {(const float*)d_in[17], (const float*)d_in[19]};
    const float* dil_pw[2]  = {(const float*)d_in[18], (const float*)d_in[20]};
    const float* w17 = (const float*)d_in[21];
    const float* w71 = (const float*)d_in[22];
    float* out = (float*)d_out;

    char* ws = (char*)d_ws; size_t off = 0;
    auto carve = [&](size_t bytes) -> void* {
        void* p = ws + off; off += (bytes + 255) & ~(size_t)255; return p;
    };
    float* xtemp = (float*)carve((size_t)NP_*4);
    float* t1    = (float*)carve((size_t)NP_*4);
    float* t2    = (float*)carve((size_t)NP_*4);
    float* acc   = (float*)carve((size_t)NP_*4);
    unsigned short* g1 = (unsigned short*)carve((size_t)NP_*2);
    unsigned short* g2 = (unsigned short*)carve((size_t)NP_*2);
    unsigned short *wb_pw1[3], *wb_pw2[3], *wb_dil[2];
    for (int s = 0; s < 3; ++s) { wb_pw1[s] = (unsigned short*)carve(CP_*CP_*2);
                                  wb_pw2[s] = (unsigned short*)carve(CP_*CP_*2); }
    for (int s = 0; s < 2; ++s)   wb_dil[s] = (unsigned short*)carve(CP_*CP_*2);
    unsigned short* wb17 = (unsigned short*)carve((size_t)7*CP_*CP_*2);
    unsigned short* wb71 = (unsigned short*)carve((size_t)7*CP_*CP_*2);
    float* avgb  = (float*)carve(B_*C_*4);
    float* maxb  = (float*)carve(B_*C_*4);
    float* ca    = (float*)carve(B_*C_*4);
    float* slist = (float*)carve(C_*4);
    int*   idxb  = (int*)carve(CP_*4);
    float* mbuf  = (float*)carve(CP_*4);
    float* vbuf  = (float*)carve(CP_*4);

    const int NPB = NP_/256, NXB = NX_/256;
    dim3 gg(HW_/64, CP_/16, B_);   // 16 x 8 x 64 WMMA macro-tiles (16M x 64N per wave)

    // weight conversions (independent)
    for (int s = 0; s < 3; ++s) {
        cvt_pw_k<<<64, 256, 0, stream>>>(sep_pw1[s], wb_pw1[s]);
        cvt_pw_k<<<64, 256, 0, stream>>>(sep_pw2[s], wb_pw2[s]);
    }
    for (int s = 0; s < 2; ++s)
        cvt_pw_k<<<64, 256, 0, stream>>>(dil_pw[s], wb_dil[s]);
    cvt_tap_k<<<448, 256, 0, stream>>>(w17, wb17);
    cvt_tap_k<<<448, 256, 0, stream>>>(w71, wb71);

    // attention -> top-k -> scale -> gather
    reduce_pool_k<<<B_*C_, 256, 0, stream>>>(x, avgb, maxb);
    attn_k<<<B_, 256, 0, stream>>>(avgb, maxb, wall, wfc1, wfc2, ca);
    slist_k<<<2, 256, 0, stream>>>(ca, slist);
    topk_k<<<1, 512, 0, stream>>>(slist, idxb);
    scale_k<<<NXB, 256, 0, stream>>>(x, ca, out);
    gather_k<<<NPB, 256, 0, stream>>>(out, idxb, xtemp);

    // identity branch (weights[3])
    acc_init_k<<<NPB, 256, 0, stream>>>(acc, xtemp, wts);

    // maxpool (weights[1]) / avgpool (weights[2]) branches
    maxpool3_k<<<NPB, 256, 0, stream>>>(xtemp, t1);
    bn_stats_k<<<CP_, 256, 0, stream>>>(t1, mbuf, vbuf);
    acc_bn_k<<<NPB, 256, 0, stream>>>(acc, t1, mbuf, vbuf, wts, 1);
    avgpool3_k<<<NPB, 256, 0, stream>>>(xtemp, t1);
    bn_stats_k<<<CP_, 256, 0, stream>>>(t1, mbuf, vbuf);
    acc_bn_k<<<NPB, 256, 0, stream>>>(acc, t1, mbuf, vbuf, wts, 2);

    // separable convs k=3,5,7 (weights[4..6])
    const int kk[3] = {3, 5, 7};
    for (int s = 0; s < 3; ++s) {
        int k = kk[s];
        dwconv_bf16_k<<<NPB, 256, 0, stream>>>(xtemp, sep_dw1[s], g1, k, 1, k/2, 1);
        pw_gemm_k<<<gg, 32, 0, stream>>>(g1, wb_pw1[s], t1);
        bn_stats_k<<<CP_, 256, 0, stream>>>(t1, mbuf, vbuf);
        bnrelu_k<<<NPB, 256, 0, stream>>>(t1, mbuf, vbuf, t2);
        dwconv_bf16_k<<<NPB, 256, 0, stream>>>(t2, sep_dw2[s], g1, k, 1, k/2, 0);
        pw_gemm_k<<<gg, 32, 0, stream>>>(g1, wb_pw2[s], t1);
        bn_stats_k<<<CP_, 256, 0, stream>>>(t1, mbuf, vbuf);
        acc_bn_k<<<NPB, 256, 0, stream>>>(acc, t1, mbuf, vbuf, wts, 4 + s);
    }

    // dilated convs k=3,5 (weights[7,8])
    for (int s = 0; s < 2; ++s) {
        int k = (s == 0) ? 3 : 5;
        dwconv_bf16_k<<<NPB, 256, 0, stream>>>(xtemp, dil_dw[s], g1, k, 2, k - 1, 1);
        pw_gemm_k<<<gg, 32, 0, stream>>>(g1, wb_dil[s], t1);
        bn_stats_k<<<CP_, 256, 0, stream>>>(t1, mbuf, vbuf);
        acc_bn_k<<<NPB, 256, 0, stream>>>(acc, t1, mbuf, vbuf, wts, 7 + s);
    }

    // 1x7 then 7x1 conv (weights[9])
    relu_cvt_k<<<NPB, 256, 0, stream>>>(xtemp, g1);
    gemm1x7_k<<<gg, 32, 0, stream>>>(g1, wb17, g2);
    gemm7x1_k<<<gg, 32, 0, stream>>>(g2, wb71, t1);
    bn_stats_k<<<CP_, 256, 0, stream>>>(t1, mbuf, vbuf);
    acc_bn_k<<<NPB, 256, 0, stream>>>(acc, t1, mbuf, vbuf, wts, 9);

    // scatter selected channels back
    scatter_k<<<NPB, 256, 0, stream>>>(acc, idxb, out);
}
